// BidirectionalAttention_20280835572117
// MI455X (gfx1250) — compile-verified, matrix operand order checked by reading, not run on hardware
//
#include <hip/hip_runtime.h>
#include <hip/hip_bf16.h>
#include <math.h>

typedef __bf16 bf16_t;
typedef __attribute__((ext_vector_type(16))) __bf16 v16bf;
typedef __attribute__((ext_vector_type(8)))  __bf16 v8bf;
typedef __attribute__((ext_vector_type(4)))  __bf16 v4bf;
typedef __attribute__((ext_vector_type(8)))  float  v8f;

#define SHUF16(lo, hi) __builtin_shufflevector((lo), (hi), 0,1,2,3,4,5,6,7,8,9,10,11,12,13,14,15)
#define WMMA_BF16(a, b, c) __builtin_amdgcn_wmma_f32_16x16x32_bf16(false, (a), false, (b), (short)0, (c), false, false)

static constexpr int B_ = 2, T_ = 2048, D_ = 2048, H_ = 16, DH_ = 128;

// Async global->LDS copy of 16 bytes per lane (ASYNCcnt-tracked; CDNA5 TDM-adjacent path).
// lds_off: low 32 bits of the generic pointer == LDS byte offset (aperture lives in addr[63:32]).
__device__ __forceinline__ void async_b128(uint32_t lds_off, const void* gaddr) {
  asm volatile("global_load_async_to_lds_b128 %0, %1, off"
               :: "v"(lds_off), "v"(gaddr)
               : "memory");
}
__device__ __forceinline__ void wait_async_le4() {
  asm volatile("s_wait_asynccnt 0x4" ::: "memory");
}
__device__ __forceinline__ void wait_async_0() {
  asm volatile("s_wait_asynccnt 0x0" ::: "memory");
}

// ---------------------------------------------------------------------------
// 0) one-shot f32 -> bf16 weight conversion (vectorized)
// ---------------------------------------------------------------------------
__global__ void __launch_bounds__(256)
convert_f32_bf16_kernel(const float* __restrict__ in, bf16_t* __restrict__ out, int n4) {
  int i = blockIdx.x * blockDim.x + threadIdx.x;
  if (i >= n4) return;
  float4 v = ((const float4*)in)[i];
  v4bf d = {(bf16_t)v.x, (bf16_t)v.y, (bf16_t)v.z, (bf16_t)v.w};
  ((v4bf*)out)[i] = d;
}

// ---------------------------------------------------------------------------
// 1) RMSNorm: h = x * rsqrt(mean(x^2)+eps) * w   (fp32 in, bf16 out)
// ---------------------------------------------------------------------------
__global__ void __launch_bounds__(256)
rmsnorm_bf16_kernel(const float* __restrict__ x, const float* __restrict__ w,
                    bf16_t* __restrict__ h, int D) {
  int row = blockIdx.x;
  const float* xr = x + (size_t)row * D;
  bf16_t* hr = h + (size_t)row * D;

  float ss = 0.f;
  const float4* x4 = (const float4*)xr;
  for (int i = threadIdx.x; i < D / 4; i += blockDim.x) {
    float4 v = x4[i];
    ss += v.x * v.x + v.y * v.y + v.z * v.z + v.w * v.w;
  }
  #pragma unroll
  for (int m = 16; m; m >>= 1) ss += __shfl_xor(ss, m, 32);

  __shared__ float red[8];
  __shared__ float invs;
  if ((threadIdx.x & 31) == 0) red[threadIdx.x >> 5] = ss;
  __syncthreads();
  if (threadIdx.x == 0) {
    float t = 0.f;
    #pragma unroll
    for (int i = 0; i < 8; i++) t += red[i];
    invs = rsqrtf(t / (float)D + 1e-6f);
  }
  __syncthreads();
  float s = invs;
  for (int i = threadIdx.x; i < D; i += blockDim.x)
    hr[i] = (bf16_t)(xr[i] * s * w[i]);
}

// ---------------------------------------------------------------------------
// 2) GEMM: C[M,N] = A[M,K](bf16) * Bw[N,K]^T(bf16), async double-buffered LDS
//    block tile 128x128, 8 waves (4x2), each wave 32x64 via 16x16x32 WMMA
// ---------------------------------------------------------------------------
template <bool RESIDUAL, typename OutT>
__global__ void __launch_bounds__(256)
gemm_bf16_wmma_kernel(const bf16_t* __restrict__ A, const bf16_t* __restrict__ Bw,
                      const float* __restrict__ resid, OutT* __restrict__ C,
                      int M, int N, int K) {
  constexpr int BM = 128, BN = 128, BK = 32;
  __shared__ __align__(16) bf16_t sA[2][BM * BK];
  __shared__ __align__(16) bf16_t sB[2][BN * BK];

  int m0 = blockIdx.y * BM;
  int n0 = blockIdx.x * BN;
  int tid = threadIdx.x;
  int wave = tid >> 5, lane = tid & 31;
  int wm = wave >> 1, wn = wave & 1;
  int half = lane >> 4, l15 = lane & 15, ko = half * 8;

  v8f zero8 = {0.f, 0.f, 0.f, 0.f, 0.f, 0.f, 0.f, 0.f};
  v8f acc[2][4];
  #pragma unroll
  for (int i = 0; i < 2; i++)
    #pragma unroll
    for (int j = 0; j < 4; j++) acc[i][j] = zero8;

  // async stage of one BK slice into buffer `buf` (4 async b128 per wave)
  auto stage_async = [&](int buf, int k0) {
    #pragma unroll
    for (int it = 0; it < 2; it++) {
      int c = tid + it * 256;
      int r = c >> 2, col = (c & 3) << 3;
      async_b128((uint32_t)(uintptr_t)&sA[buf][r * BK + col],
                 &A[(size_t)(m0 + r) * K + k0 + col]);
    }
    #pragma unroll
    for (int it = 0; it < 2; it++) {
      int c = tid + it * 256;
      int r = c >> 2, col = (c & 3) << 3;
      async_b128((uint32_t)(uintptr_t)&sB[buf][r * BK + col],
                 &Bw[(size_t)(n0 + r) * K + k0 + col]);
    }
  };

  stage_async(0, 0);
  int nsteps = K / BK;
  for (int s = 0; s < nsteps; s++) {
    int buf = s & 1;
    if (s + 1 < nsteps) {
      stage_async(buf ^ 1, (s + 1) * BK);
      wait_async_le4();   // oldest 4 (this buffer) complete; prefetch stays in flight
    } else {
      wait_async_0();
    }
    __syncthreads();

    v16bf af[2], bf[4];
    #pragma unroll
    for (int i = 0; i < 2; i++) {
      const bf16_t* p = &sA[buf][(wm * 32 + i * 16 + l15) * BK];
      v8bf lo = *(const v8bf*)(p + ko);
      v8bf hi = *(const v8bf*)(p + 16 + ko);
      af[i] = SHUF16(lo, hi);
    }
    #pragma unroll
    for (int j = 0; j < 4; j++) {
      const bf16_t* p = &sB[buf][(wn * 64 + j * 16 + l15) * BK];
      v8bf lo = *(const v8bf*)(p + ko);
      v8bf hi = *(const v8bf*)(p + 16 + ko);
      bf[j] = SHUF16(lo, hi);
    }
    #pragma unroll
    for (int i = 0; i < 2; i++)
      #pragma unroll
      for (int j = 0; j < 4; j++)
        acc[i][j] = WMMA_BF16(af[i], bf[j], acc[i][j]);

    __syncthreads(); // all waves done reading buf before it is overwritten
  }

  #pragma unroll
  for (int i = 0; i < 2; i++)
    #pragma unroll
    for (int j = 0; j < 4; j++)
      #pragma unroll
      for (int r = 0; r < 8; r++) {
        int row = m0 + wm * 32 + i * 16 + r + half * 8;
        int col = n0 + wn * 64 + j * 16 + l15;
        float v = acc[i][j][r];
        if constexpr (RESIDUAL) v += resid[(size_t)row * N + col];
        C[(size_t)row * N + col] = (OutT)v;
      }
}

// ---------------------------------------------------------------------------
// 3) RoPE + split: qkv(bf16, [B*T][3*D]) -> q,k ([B*H][T][128]) and V^T
// ---------------------------------------------------------------------------
__global__ void __launch_bounds__(256)
rope_split_kernel(const bf16_t* __restrict__ qkv, bf16_t* __restrict__ q,
                  bf16_t* __restrict__ k, bf16_t* __restrict__ vT) {
  int idx = blockIdx.x * blockDim.x + threadIdx.x;
  const int total = B_ * H_ * T_ * (DH_ / 2);
  if (idx >= total) return;
  int d = idx & 63;
  int t = (idx >> 6) & (T_ - 1);
  int h = (idx >> 17) & (H_ - 1);
  int b = idx >> 21;

  size_t row = (size_t)(b * T_ + t) * (3 * D_);
  int cq = 0 * D_ + h * DH_ + d;
  int ck = 1 * D_ + h * DH_ + d;
  int cv = 2 * D_ + h * DH_ + d;
  float q1 = (float)qkv[row + cq], q2 = (float)qkv[row + cq + 64];
  float k1 = (float)qkv[row + ck], k2 = (float)qkv[row + ck + 64];
  float v1 = (float)qkv[row + cv], v2 = (float)qkv[row + cv + 64];

  float invf = powf(10000.f, -(float)d / 64.f);
  float ang = (float)t * invf;
  float c = __cosf(ang), s = __sinf(ang);

  size_t bh = (size_t)(b * H_ + h);
  size_t o = (bh * T_ + t) * DH_ + d;
  q[o]      = (bf16_t)(q1 * c - q2 * s);
  q[o + 64] = (bf16_t)(q2 * c + q1 * s);
  k[o]      = (bf16_t)(k1 * c - k2 * s);
  k[o + 64] = (bf16_t)(k2 * c + k1 * s);
  size_t ov = (bh * DH_ + d) * (size_t)T_ + t;
  vT[ov]            = (bf16_t)v1;
  vT[ov + 64 * T_]  = (bf16_t)v2;
}

// ---------------------------------------------------------------------------
// 4) Flash attention: per (b,h), 128-query tile, 64-key tiles, bf16 WMMA,
//    K / V^T tiles staged with async global->LDS copies
// ---------------------------------------------------------------------------
__global__ void __launch_bounds__(256)
flash_attn_kernel(const bf16_t* __restrict__ q, const bf16_t* __restrict__ k,
                  const bf16_t* __restrict__ vT, bf16_t* __restrict__ o) {
  constexpr int BQ = 128, BKEY = 64;
  __shared__ __align__(16) bf16_t sK[BKEY * DH_];    // 16 KB
  __shared__ __align__(16) bf16_t sVT[DH_ * BKEY];   // 16 KB
  __shared__ __align__(16) bf16_t sP[8 * 16 * BKEY]; // 16 KB

  int bh = blockIdx.y;    // b*H + h
  int qtile = blockIdx.x; // 0..T/BQ-1
  int tid = threadIdx.x, wave = tid >> 5, lane = tid & 31;
  int half = lane >> 4, l15 = lane & 15, ko = half * 8;

  // per-wave 16 query rows, Q fragments held in registers (4 k-steps of 32)
  const bf16_t* qbase = q + ((size_t)bh * T_ + (size_t)qtile * BQ + wave * 16) * DH_;
  v16bf qf[4];
  #pragma unroll
  for (int s = 0; s < 4; s++) {
    const bf16_t* p = qbase + (size_t)l15 * DH_ + s * 32;
    v8bf lo = *(const v8bf*)(p + ko);
    v8bf hi = *(const v8bf*)(p + 16 + ko);
    qf[s] = SHUF16(lo, hi);
  }

  v8f zero8 = {0.f, 0.f, 0.f, 0.f, 0.f, 0.f, 0.f, 0.f};
  v8f accO[8];
  float mrow[8], lrow[8];
  #pragma unroll
  for (int n = 0; n < 8; n++) accO[n] = zero8;
  #pragma unroll
  for (int r = 0; r < 8; r++) { mrow[r] = -1e30f; lrow[r] = 0.f; }

  const float scale = 0.0883883476483184f; // 1/sqrt(128)
  bf16_t* pw = &sP[wave * 16 * BKEY];

  for (int key0 = 0; key0 < T_; key0 += BKEY) {
    __syncthreads(); // previous iteration's readers done before overwrite
    // stage K tile (64 x 128) via async copies
    const bf16_t* kg = k + ((size_t)bh * T_ + key0) * DH_;
    #pragma unroll
    for (int it = 0; it < 4; it++) {
      int c = tid + it * 256;
      int r = c >> 4, col = (c & 15) << 3;
      async_b128((uint32_t)(uintptr_t)&sK[r * DH_ + col], &kg[(size_t)r * DH_ + col]);
    }
    // stage V^T tile (128 x 64) via async copies
    const bf16_t* vg = vT + (size_t)bh * DH_ * T_ + key0;
    #pragma unroll
    for (int it = 0; it < 4; it++) {
      int c = tid + it * 256;
      int r = c >> 3, col = (c & 7) << 3;
      async_b128((uint32_t)(uintptr_t)&sVT[r * BKEY + col], &vg[(size_t)r * T_ + col]);
    }
    wait_async_0();
    __syncthreads();

    // S = Q K^T (16 x 64 per wave)
    v8f accS[4];
    #pragma unroll
    for (int t = 0; t < 4; t++) accS[t] = zero8;
    #pragma unroll
    for (int t = 0; t < 4; t++)
      #pragma unroll
      for (int s = 0; s < 4; s++) {
        const bf16_t* p = &sK[(t * 16 + l15) * DH_ + s * 32];
        v8bf lo = *(const v8bf*)(p + ko);
        v8bf hi = *(const v8bf*)(p + 16 + ko);
        v16bf bk = SHUF16(lo, hi);
        accS[t] = WMMA_BF16(qf[s], bk, accS[t]);
      }

    // online softmax (row stats across 16-lane groups; rows M=r+8*half)
    float pm[8];
    #pragma unroll
    for (int r = 0; r < 8; r++) {
      float mx = fmaxf(fmaxf(accS[0][r], accS[1][r]), fmaxf(accS[2][r], accS[3][r]));
      mx *= scale;
      mx = fmaxf(mx, __shfl_xor(mx, 1, 32));
      mx = fmaxf(mx, __shfl_xor(mx, 2, 32));
      mx = fmaxf(mx, __shfl_xor(mx, 4, 32));
      mx = fmaxf(mx, __shfl_xor(mx, 8, 32));
      pm[r] = mx;
    }
    float alpha[8];
    #pragma unroll
    for (int r = 0; r < 8; r++) {
      float mnew = fmaxf(mrow[r], pm[r]);
      alpha[r] = __expf(mrow[r] - mnew);
      mrow[r] = mnew;
    }
    float rs[8];
    #pragma unroll
    for (int r = 0; r < 8; r++) rs[r] = 0.f;
    #pragma unroll
    for (int t = 0; t < 4; t++)
      #pragma unroll
      for (int r = 0; r < 8; r++) {
        float p = __expf(accS[t][r] * scale - mrow[r]);
        rs[r] += p;
        pw[(r + half * 8) * BKEY + t * 16 + l15] = (bf16_t)p;
      }
    #pragma unroll
    for (int r = 0; r < 8; r++) {
      float s = rs[r];
      s += __shfl_xor(s, 1, 32);
      s += __shfl_xor(s, 2, 32);
      s += __shfl_xor(s, 4, 32);
      s += __shfl_xor(s, 8, 32);
      lrow[r] = lrow[r] * alpha[r] + s;
    }
    #pragma unroll
    for (int n = 0; n < 8; n++)
      #pragma unroll
      for (int r = 0; r < 8; r++) accO[n][r] *= alpha[r];
    __syncthreads(); // make P visible / keep waves in lockstep

    // O += P V  (P as A-fragments from LDS, V^T rows as B-fragments)
    #pragma unroll
    for (int s = 0; s < 2; s++) {
      const bf16_t* pp = pw + (size_t)l15 * BKEY + s * 32;
      v8bf plo = *(const v8bf*)(pp + ko);
      v8bf phi = *(const v8bf*)(pp + 16 + ko);
      v16bf ap = SHUF16(plo, phi);
      #pragma unroll
      for (int n = 0; n < 8; n++) {
        const bf16_t* vp = &sVT[(n * 16 + l15) * BKEY + s * 32];
        v8bf vlo = *(const v8bf*)(vp + ko);
        v8bf vhi = *(const v8bf*)(vp + 16 + ko);
        v16bf bv = SHUF16(vlo, vhi);
        accO[n] = WMMA_BF16(ap, bv, accO[n]);
      }
    }
  }

  // epilogue: O / l, store to (B*T, D) layout col = h*128 + dh
  int b = bh >> 4, h = bh & 15;
  #pragma unroll
  for (int r = 0; r < 8; r++) {
    float inv = 1.f / lrow[r];
    int trow = qtile * BQ + wave * 16 + r + half * 8;
    size_t orow = (size_t)(b * T_ + trow) * D_ + h * DH_;
    #pragma unroll
    for (int n = 0; n < 8; n++)
      o[orow + n * 16 + l15] = (bf16_t)(accO[n][r] * inv);
  }
}

// ---------------------------------------------------------------------------
extern "C" void kernel_launch(void* const* d_in, const int* in_sizes, int n_in,
                              void* d_out, int out_size, void* d_ws, size_t ws_size,
                              hipStream_t stream) {
  const float* x     = (const float*)d_in[0];
  const float* normw = (const float*)d_in[1];
  const float* wqkv  = (const float*)d_in[2];
  const float* wout  = (const float*)d_in[3];
  float* out = (float*)d_out;

  char* ws = (char*)d_ws;
  size_t off = 0;
  bf16_t* h_bf    = (bf16_t*)(ws + off); off += (size_t)B_ * T_ * D_ * 2;       // 16.8 MB
  bf16_t* qkv     = (bf16_t*)(ws + off); off += (size_t)B_ * T_ * 3 * D_ * 2;   // 50.3 MB
  bf16_t* qb      = (bf16_t*)(ws + off); off += (size_t)B_ * H_ * T_ * DH_ * 2; // 16.8 MB
  bf16_t* kb      = (bf16_t*)(ws + off); off += (size_t)B_ * H_ * T_ * DH_ * 2;
  bf16_t* vTb     = (bf16_t*)(ws + off); off += (size_t)B_ * H_ * T_ * DH_ * 2;
  bf16_t* attn    = (bf16_t*)(ws + off); off += (size_t)B_ * T_ * D_ * 2;
  bf16_t* wqkv_bf = (bf16_t*)(ws + off); off += (size_t)3 * D_ * D_ * 2;        // 25.2 MB
  bf16_t* wout_bf = (bf16_t*)(ws + off); off += (size_t)D_ * D_ * 2;            //  8.4 MB

  // 0) weight conversion (once per launch)
  {
    int n4 = 3 * D_ * D_ / 4;
    convert_f32_bf16_kernel<<<(n4 + 255) / 256, 256, 0, stream>>>(wqkv, wqkv_bf, n4);
    n4 = D_ * D_ / 4;
    convert_f32_bf16_kernel<<<(n4 + 255) / 256, 256, 0, stream>>>(wout, wout_bf, n4);
  }

  // 1) RMSNorm -> bf16 h
  rmsnorm_bf16_kernel<<<B_ * T_, 256, 0, stream>>>(x, normw, h_bf, D_);

  // 2) QKV projection: (4096 x 2048) @ (6144 x 2048)^T -> bf16 qkv
  dim3 g1(3 * D_ / 128, (B_ * T_) / 128);
  gemm_bf16_wmma_kernel<false, bf16_t><<<g1, 256, 0, stream>>>(
      h_bf, wqkv_bf, nullptr, qkv, B_ * T_, 3 * D_, D_);

  // 3) RoPE + split into q, k, V^T
  int total = B_ * H_ * T_ * (DH_ / 2);
  rope_split_kernel<<<(total + 255) / 256, 256, 0, stream>>>(qkv, qb, kb, vTb);

  // 4) attention
  dim3 g2(T_ / 128, B_ * H_);
  flash_attn_kernel<<<g2, 256, 0, stream>>>(qb, kb, vTb, attn);

  // 5) output projection + residual -> fp32 out
  dim3 g3(D_ / 128, (B_ * T_) / 128);
  gemm_bf16_wmma_kernel<true, float><<<g3, 256, 0, stream>>>(
      attn, wout_bf, x, out, B_ * T_, D_, D_);
}
